// PCENetwork_42717744726513
// MI455X (gfx1250) — matrix operand order repaired and checked by reading, not compiled, not run on hardware
//
#include <hip/hip_runtime.h>
#include <hip/hip_bf16.h>

// ---------------------------------------------------------------------------
// CDNA5 (gfx1250) implementation of the patch-MoE conv network.
// Expert 3x3 convs (98% of FLOPs) run as implicit GEMM on v_wmma_f32_16x16x32_f16,
// with an LDS-staged im2col chunk so B fragments load as ds_load_b128 pairs.
// ---------------------------------------------------------------------------

typedef _Float16 v8h  __attribute__((ext_vector_type(8)));
typedef _Float16 v16h __attribute__((ext_vector_type(16)));
typedef float    v8f  __attribute__((ext_vector_type(8)));

__device__ __forceinline__ float posval(int pc, int Y, int X, int H, int W) {
  float gx = (float)X / (float)(W - 1);
  float gy = (float)Y / (float)(H - 1);
  if (pc == 0) return gx;
  if (pc == 1) return gy;
  if (pc == 2) return __sinf(3.14159265358979323f * gx);
  return __sinf(3.14159265358979323f * gy);
}

// ---------------------------------------------------------------------------
// Weight conversion: (E*Co, K) f32 -> (E*Co, KPAD) f16, zero padded in K.
// ---------------------------------------------------------------------------
__global__ void wconv_kernel(const float* __restrict__ ew, _Float16* __restrict__ wf,
                             int ECo, int K, int KPAD) {
  int i = blockIdx.x * blockDim.x + threadIdx.x;
  if (i >= ECo * KPAD) return;
  int row = i / KPAD, k = i - row * KPAD;
  wf[i] = (k < K) ? (_Float16)ew[(size_t)row * K + k] : (_Float16)0.f;
}

__global__ void initmax_kernel(int* mb) {
  if (threadIdx.x < 4) mb[threadIdx.x] = 0;
}

__global__ void gate_kernel(const int* __restrict__ mb, float* __restrict__ gate) {
  if (threadIdx.x < 4)
    gate[threadIdx.x] = (__int_as_float(mb[threadIdx.x]) > 0.2f) ? 1.f : 0.f;
}

// ---------------------------------------------------------------------------
// Router: per-patch 3x3 proj conv (pad=1 inside patch) -> global avg pool ->
// linear -> softmax; also global per-expert max of scores (atomicMax on bits).
// One 64-thread block per patch (one thread per pixel).
// ---------------------------------------------------------------------------
template <int CRAW, int C, int H, int W>
__global__ void router_kernel(const float* __restrict__ X,
                              const float* __restrict__ pw,   // (8,C,3,3)
                              const float* __restrict__ pb,   // (8)
                              const float* __restrict__ rw,   // (4,8)
                              const float* __restrict__ rb,   // (4)
                              float* __restrict__ scores,     // (BP,4)
                              int* __restrict__ maxbits) {    // (4)
  constexpr int PR = W / 8;
  constexpr int P  = (H / 8) * PR;
  __shared__ _Float16 sP[C * 100];     // (C, 10, 10) zero border
  __shared__ float sRed[8 * 64];
  const int t  = threadIdx.x;          // 64 threads
  const int pl = blockIdx.x;           // b*P + p
  const int b = pl / P, pp = pl - b * P, py = pp / PR, px = pp - py * PR;

  for (int i = t; i < C * 100; i += 64) sP[i] = (_Float16)0.f;
  __syncthreads();
  for (int i = t; i < C * 64; i += 64) {
    int ci = i >> 6, r = i & 63, y = r >> 3, x = r & 7;
    int Y = py * 8 + y, Xg = px * 8 + x;
    float v = (ci < CRAW) ? X[(((size_t)b * CRAW + ci) * H + Y) * W + Xg]
                          : posval(ci - CRAW, Y, Xg, H, W);
    sP[ci * 100 + (y + 1) * 10 + (x + 1)] = (_Float16)v;
  }
  __syncthreads();

  const int y = t >> 3, x = t & 7;
  float acc[8];
#pragma unroll
  for (int oc = 0; oc < 8; ++oc) acc[oc] = 0.f;
  for (int ci = 0; ci < C; ++ci)
#pragma unroll
    for (int ky = 0; ky < 3; ++ky)
#pragma unroll
      for (int kx = 0; kx < 3; ++kx) {
        float v = (float)sP[ci * 100 + (y + ky) * 10 + (x + kx)];
        const float* wp = pw + (ci * 3 + ky) * 3 + kx;
#pragma unroll
        for (int oc = 0; oc < 8; ++oc) acc[oc] += wp[oc * C * 9] * v;
      }
#pragma unroll
  for (int oc = 0; oc < 8; ++oc) sRed[oc * 64 + t] = acc[oc];
  __syncthreads();
  for (int s = 32; s > 0; s >>= 1) {
    if (t < s)
#pragma unroll
      for (int oc = 0; oc < 8; ++oc)
        sRed[oc * 64 + t] += sRed[oc * 64 + t + s];
    __syncthreads();
  }
  if (t == 0) {
    float lg[4], mx = -1e30f;
#pragma unroll
    for (int e = 0; e < 4; ++e) {
      float a = rb[e];
#pragma unroll
      for (int oc = 0; oc < 8; ++oc) {
        float pooled = pb[oc] + sRed[oc * 64] * (1.f / 64.f);
        a += rw[e * 8 + oc] * pooled;
      }
      lg[e] = a;
      mx = fmaxf(mx, a);
    }
    float ssum = 0.f;
#pragma unroll
    for (int e = 0; e < 4; ++e) { lg[e] = __expf(lg[e] - mx); ssum += lg[e]; }
    float inv = 1.f / ssum;
#pragma unroll
    for (int e = 0; e < 4; ++e) {
      float s = lg[e] * inv;
      scores[pl * 4 + e] = s;
      atomicMax(&maxbits[e], __float_as_int(s));   // s > 0 -> bit order == float order
    }
  }
}

// ---------------------------------------------------------------------------
// Experts as implicit GEMM with WMMA.
//   M = 4*CO weight rows (f16, KPAD-padded), K = C*9 (padded), N = 64 pixels.
// Block = 256 threads (8 wave32), one patch per block; blockIdx.y = 32-co slice.
// Wave w: expert e = w&3, co-half ch = w>>2 (16 channels), 4 N-tiles => 4 accum
// tiles per wave. The im2col B matrix is staged per-256-K chunk in LDS in
// fragment-native [pixel][k] layout (row stride padded to 264 halves so the
// b128 lane reads per fragment hit disjoint bank groups). Per kb step all four
// B fragments are loaded first (8 ds_load_b128 in flight), then the 4 WMMAs
// issue back-to-back on the accumulate chain. Cross-expert combine
// (score * relu) goes through an LDS buffer re-using the chunk storage.
// A frag: lane (l&15) = row, (l>>4) selects K halves {0..15}/{16..31}.
// B frag: lanes 0-15 hold K rows 0..15, lanes 16-31 hold 16..31; col = l&15.
// C/D  : VGPR r -> m = 8*(l>>4) + r, pixel n = tile*16 + (l&15).
// ---------------------------------------------------------------------------
template <int CRAW, int C, int CPAD, int CO, int KPAD, int H, int W>
__global__ void __launch_bounds__(256)
expert_kernel(const float* __restrict__ X,
              const _Float16* __restrict__ Wf,  // (4*CO, KPAD)
              const float* __restrict__ eb,     // (4*CO)
              const float* __restrict__ scores, // (BP,4)
              const float* __restrict__ gate,   // (4)
              float* __restrict__ comb) {       // (B,CO,H,W)
  constexpr int PR = W / 8;
  constexpr int P  = (H / 8) * PR;
  constexpr int KCHUNK  = 256;
  constexpr int BSTRIDE = KCHUNK + 8;   // 528B rows -> conflict-free b128 reads

  union ExpSh {
    _Float16 bt[64 * BSTRIDE];          // im2col chunk  [pixel][k]
    float    cmb[8 * 16 * 64];          // per-(e,ch) scaled expert outputs
  };
  __shared__ _Float16 sP[CPAD * 100];   // zero-padded 10x10 patch, padded channels
  __shared__ ExpSh su;

  const int t    = threadIdx.x;         // 256
  const int lane = t & 31;
  const int wave = t >> 5;
  const int e    = wave & 3;            // expert for this wave
  const int ch   = wave >> 2;           // which 16-channel half of the co slice
  const int lm   = lane & 15;
  const int hi   = lane >> 4;
  const int pl   = blockIdx.x;
  const int b = pl / P, pp = pl - b * P, py = pp / PR, px = pp - py * PR;
  const int coBase = blockIdx.y * 32;

  // ---- stage patch into LDS (f16, zero border, zero padded channels) ----
  for (int i = t; i < CPAD * 100; i += 256) sP[i] = (_Float16)0.f;
  __syncthreads();
  for (int i = t; i < C * 64; i += 256) {
    int ci = i >> 6, r = i & 63, y = r >> 3, x = r & 7;
    int Y = py * 8 + y, Xg = px * 8 + x;
    float v = (ci < CRAW) ? X[(((size_t)b * CRAW + ci) * H + Y) * W + Xg]
                          : posval(ci - CRAW, Y, Xg, H, W);
    sP[ci * 100 + (y + 1) * 10 + (x + 1)] = (_Float16)v;
  }

  v8f acc[4];
#pragma unroll
  for (int nt = 0; nt < 4; ++nt)
#pragma unroll
    for (int j = 0; j < 8; ++j) acc[nt][j] = 0.f;

  const _Float16* wrow =
      Wf + (size_t)(e * CO + coBase + ch * 16 + lm) * KPAD + hi * 16;

  for (int kc = 0; kc < KPAD; kc += KCHUNK) {
    const int kLim = (KPAD - kc < KCHUNK) ? (KPAD - kc) : KCHUNK;  // multiple of 32
    const int nv   = kLim >> 3;       // v8h groups per pixel row
    __syncthreads();                  // sP ready / previous chunk consumed
    // ---- cooperative im2col build: bt[n][kk] for kk in [0,kLim) ----
    for (int idx = t; idx < 64 * nv; idx += 256) {
      int n = idx / nv, blk = idx - n * nv;
      int yy = n >> 3, xx = n & 7;
      int k0 = kc + blk * 8;
      v8h vv;
#pragma unroll
      for (int j = 0; j < 8; ++j) {
        int k  = k0 + j;
        int ci = k / 9, rr = k - ci * 9;
        int ky = rr / 3, kx = rr - ky * 3;
        vv[j] = sP[ci * 100 + (yy + ky) * 10 + (xx + kx)];
      }
      *(v8h*)&su.bt[n * BSTRIDE + blk * 8] = vv;
    }
    __syncthreads();
    if (kc + KCHUNK < KPAD)
      __builtin_prefetch(wrow + kc + KCHUNK, 0, 1);   // global_prefetch_b8
    // ---- WMMA over the chunk: load A + all 4 B fragments, then 4 MMAs ----
#pragma unroll 2
    for (int kb = 0; kb < (kLim >> 5); ++kb) {
      v8h alo = *(const v8h*)(wrow + kc + kb * 32);
      v8h ahi = *(const v8h*)(wrow + kc + kb * 32 + 8);
      v16h a  = __builtin_shufflevector(alo, ahi, 0, 1, 2, 3, 4, 5, 6, 7,
                                        8, 9, 10, 11, 12, 13, 14, 15);
      v16h bf[4];
#pragma unroll
      for (int nt = 0; nt < 4; ++nt) {
        const _Float16* bp = &su.bt[(nt * 16 + lm) * BSTRIDE + kb * 32 + hi * 16];
        v8h blo = *(const v8h*)bp;        // ds_load_b128
        v8h bhi = *(const v8h*)(bp + 8);  // ds_load_b128
        bf[nt] = __builtin_shufflevector(blo, bhi, 0, 1, 2, 3, 4, 5, 6, 7,
                                         8, 9, 10, 11, 12, 13, 14, 15);
      }
#pragma unroll
      for (int nt = 0; nt < 4; ++nt)
        acc[nt] = __builtin_amdgcn_wmma_f32_16x16x32_f16(
            false, a, false, bf[nt], (short)0, acc[nt], false, false);
    }
  }

  // ---- epilogue: bias + relu + score scale per expert wave, LDS combine ----
  const float s = scores[pl * 4 + e] * gate[e];
  __syncthreads();                      // all waves done reading bt
#pragma unroll
  for (int nt = 0; nt < 4; ++nt)
#pragma unroll
    for (int r = 0; r < 8; ++r) {
      int m  = hi * 8 + r;
      int co = coBase + ch * 16 + m;
      float v = acc[nt][r] + eb[e * CO + co];
      su.cmb[((e * 2 + ch) * 16 + m) * 64 + nt * 16 + lm] = s * fmaxf(v, 0.f);
    }
  __syncthreads();
  for (int i = t; i < 2 * 16 * 64; i += 256) {
    int n = i & 63, m = (i >> 6) & 15, c2 = i >> 10;
    float v = su.cmb[((0 * 2 + c2) * 16 + m) * 64 + n]
            + su.cmb[((1 * 2 + c2) * 16 + m) * 64 + n]
            + su.cmb[((2 * 2 + c2) * 16 + m) * 64 + n]
            + su.cmb[((3 * 2 + c2) * 16 + m) * 64 + n];
    int co = coBase + c2 * 16 + m;
    int yy = n >> 3, xx = n & 7;
    comb[(((size_t)b * CO + co) * H + (py * 8 + yy)) * W + (px * 8 + xx)] = v;
  }
}

// ---------------------------------------------------------------------------
// 1x1 conv with padding 4 (output grows by 8); border pixels get bias only.
// ---------------------------------------------------------------------------
__global__ void fin_kernel(const float* __restrict__ comb, const float* __restrict__ fw,
                           const float* __restrict__ fb, float* __restrict__ out,
                           int B, int CO, int Hin, int Win) {
  const int Hout = Hin + 8, Wout = Win + 8;
  const long total = (long)B * CO * Hout * Wout;
  long i = (long)blockIdx.x * blockDim.x + threadIdx.x;
  if (i >= total) return;
  int Xo = (int)(i % Wout); long r1 = i / Wout;
  int Yo = (int)(r1 % Hout); long r2 = r1 / Hout;
  int co = (int)(r2 % CO);  int b = (int)(r2 / CO);
  float acc = fb[co];
  int Yi = Yo - 4, Xi = Xo - 4;
  if (Yi >= 0 && Yi < Hin && Xi >= 0 && Xi < Win) {
    const float* cp = comb + (((size_t)b * CO) * Hin + Yi) * Win + Xi;
    const float* wp = fw + (size_t)co * CO;
    const size_t stride = (size_t)Hin * Win;
    for (int ci = 0; ci < CO; ++ci) acc += wp[ci] * cp[(size_t)ci * stride];
  }
  out[i] = acc;
}

// ---------------------------------------------------------------------------
// Final linear: (16, feat) @ lin_w.T + lin_b. Split-K + LDS reduce + atomicAdd.
// ---------------------------------------------------------------------------
__global__ void out_init_kernel(const float* __restrict__ lb, float* __restrict__ out) {
  int i = threadIdx.x;
  if (i < 160) out[i] = lb[i % 10];
}

__global__ void linear_kernel(const float* __restrict__ Xf, const float* __restrict__ lw,
                              float* __restrict__ out, int feat, int chunk) {
  const int b  = blockIdx.y;
  const int k0 = blockIdx.x * chunk;
  __shared__ float sRed[10 * 256];
  float acc[10];
#pragma unroll
  for (int o = 0; o < 10; ++o) acc[o] = 0.f;
  for (int k = k0 + threadIdx.x; k < k0 + chunk; k += 256) {
    float x = Xf[(size_t)b * feat + k];
#pragma unroll
    for (int o = 0; o < 10; ++o) acc[o] += x * lw[(size_t)o * feat + k];
  }
#pragma unroll
  for (int o = 0; o < 10; ++o) sRed[o * 256 + threadIdx.x] = acc[o];
  __syncthreads();
  for (int s = 128; s > 0; s >>= 1) {
    if (threadIdx.x < s)
#pragma unroll
      for (int o = 0; o < 10; ++o)
        sRed[o * 256 + threadIdx.x] += sRed[o * 256 + threadIdx.x + s];
    __syncthreads();
  }
  if (threadIdx.x < 10) atomicAdd(&out[b * 10 + threadIdx.x], sRed[threadIdx.x * 256]);
}

// ---------------------------------------------------------------------------
// Host-side layer driver
// ---------------------------------------------------------------------------
template <int CRAW, int C, int CPAD, int CO, int KPAD, int H, int W>
static void run_layer(const float* Xin, const float* pw, const float* pb,
                      const float* ew, const float* ebias,
                      const float* fw, const float* fb,
                      const float* rw, const float* rb,
                      float* comb, float* Xout, _Float16* wf16,
                      float* scores, int* maxb, float* gate, hipStream_t stream) {
  constexpr int P  = (H / 8) * (W / 8);
  constexpr int BP = 16 * P;
  constexpr int K  = C * 9;
  initmax_kernel<<<1, 4, 0, stream>>>(maxb);
  const int wtotal = 4 * CO * KPAD;
  wconv_kernel<<<(wtotal + 255) / 256, 256, 0, stream>>>(ew, wf16, 4 * CO, K, KPAD);
  router_kernel<CRAW, C, H, W><<<BP, 64, 0, stream>>>(Xin, pw, pb, rw, rb, scores, maxb);
  gate_kernel<<<1, 4, 0, stream>>>(maxb, gate);
  expert_kernel<CRAW, C, CPAD, CO, KPAD, H, W>
      <<<dim3(BP, CO / 32), 256, 0, stream>>>(Xin, wf16, ebias, scores, gate, comb);
  const long ftotal = 16L * CO * (H + 8) * (W + 8);
  fin_kernel<<<(int)((ftotal + 255) / 256), 256, 0, stream>>>(comb, fw, fb, Xout,
                                                              16, CO, H, W);
}

extern "C" void kernel_launch(void* const* d_in, const int* in_sizes, int n_in,
                              void* d_out, int out_size, void* d_ws, size_t ws_size,
                              hipStream_t stream) {
  (void)in_sizes; (void)n_in; (void)out_size; (void)ws_size;
  auto in = [&](int i) { return (const float*)d_in[i]; };
  const float* X  = in(0);
  const float* rw = in(25);
  const float* rb = in(26);
  const float* lw = in(27);
  const float* lb = in(28);

  // Workspace layout (bytes, 256-aligned); total ~204 MB.
  char* ws = (char*)d_ws;
  const size_t szA = 16ull * 128 * 88 * 88 * 4;  // 63,438,848
  const size_t szB = 16ull * 128 * 96 * 96 * 4;  // 75,497,472
  const size_t szC = szA;                        // comb scratch (max layer-3)
  const size_t szW = 512ull * 1216 * 2;          // padded f16 weights
  const size_t szS = 1936ull * 4 * 4;            // scores
  float*    bufA   = (float*)(ws);
  float*    bufB   = (float*)(ws + szA);
  float*    comb   = (float*)(ws + szA + szB);
  _Float16* wf16   = (_Float16*)(ws + szA + szB + szC);
  float*    scores = (float*)(ws + szA + szB + szC + szW);
  int*      maxb   = (int*)(ws + szA + szB + szC + szW + szS);
  float*    gate   = (float*)(ws + szA + szB + szC + szW + szS + 256);

  // Layer 0: (16,3,64,64) -> (16,64,72,72)
  run_layer<3, 7, 8, 64, 64, 64, 64>(X, in(1), in(2), in(3), in(4), in(5), in(6),
                                     rw, rb, comb, bufA, wf16, scores, maxb, gate, stream);
  // Layer 1: (16,64,72,72) -> (16,64,80,80)
  run_layer<64, 68, 72, 64, 640, 72, 72>(bufA, in(7), in(8), in(9), in(10), in(11), in(12),
                                         rw, rb, comb, bufB, wf16, scores, maxb, gate, stream);
  // Layer 2: (16,64,80,80) -> (16,128,88,88)
  run_layer<64, 68, 72, 128, 640, 80, 80>(bufB, in(13), in(14), in(15), in(16), in(17), in(18),
                                          rw, rb, comb, bufA, wf16, scores, maxb, gate, stream);
  // Layer 3: (16,128,88,88) -> (16,128,96,96)
  run_layer<128, 132, 136, 128, 1216, 88, 88>(bufA, in(19), in(20), in(21), in(22), in(23), in(24),
                                              rw, rb, comb, bufB, wf16, scores, maxb, gate, stream);

  // Final linear: feat = 128*96*96 = 1179648 ; 144 K-chunks of 8192.
  out_init_kernel<<<1, 256, 0, stream>>>(lb, (float*)d_out);
  linear_kernel<<<dim3(144, 16), 256, 0, stream>>>(bufB, lw, (float*)d_out, 1179648, 8192);
}
